// AutoCorrelation_35871566856387
// MI455X (gfx1250) — matrix operand order repaired and checked by reading, not compiled, not run on hardware
//
#include <hip/hip_runtime.h>

// ---------------------------------------------------------------------------
// AutoCorrelation (Autoformer) for MI455X / gfx1250.
// corr via four-step matmul FFT (2048 = 64 x 32) on v_wmma_f32_16x16x32_bf16
// with split-bf16 (hi+lo) 3-term products for ~f32-class accuracy, fused with
// per-position top-7 over channels, softmax, and delayed-V gather.
// ---------------------------------------------------------------------------

#define AC_B    16
#define AC_L    2048
#define AC_E    512
#define AC_H    8
#define AC_HD   64
#define AC_TOPK 7

typedef __attribute__((ext_vector_type(16))) __bf16 v16bf;
typedef __attribute__((ext_vector_type(8)))  float  v8f;

union V16U { v16bf v; unsigned short u[16]; };

static __device__ inline unsigned short f2bf(float f) {
  union { float f; unsigned int u; } x; x.f = f;
  unsigned int u = x.u;
  unsigned int lsb = (u >> 16) & 1u;
  u += 0x7fffu + lsb;                 // round-to-nearest-even
  return (unsigned short)(u >> 16);
}
static __device__ inline float bf2f(unsigned short b) {
  union { unsigned int u; float f; } x; x.u = ((unsigned int)b) << 16;
  return x.f;
}

static __device__ inline v8f wmma_bf16(v16bf a, v16bf b, v8f c) {
  // D = A(16x32) * B(32x16) + C, f32 accumulate
  return __builtin_amdgcn_wmma_f32_16x16x32_bf16(false, a, false, b, (short)0, c, false, false);
}
// split-bf16 product: A*B ~= Ah*Bh + Ah*Bl + Al*Bh  (lo*lo term ~2^-18, dropped)
static __device__ inline v8f wmma3(v16bf aH, v16bf aL, v16bf bH, v16bf bL, v8f c) {
  c = wmma_bf16(aH, bH, c);
  c = wmma_bf16(aH, bL, c);
  c = wmma_bf16(aL, bH, c);
  return c;
}

__global__ __launch_bounds__(256, 1)
void autocorr_fft_topk_kernel(const float* __restrict__ Qg,
                              const float* __restrict__ Kg,
                              const float* __restrict__ Vg,
                              float* __restrict__ Og)
{
  const int tid  = threadIdx.x;
  const int lane = tid & 31;          // wave32
  const int wave = tid >> 5;          // 8 waves
  const int bh   = blockIdx.x;
  const int b    = bh >> 3;
  const int h    = bh & 7;

  // ---------------- LDS carve (194560 bytes total) ----------------
  extern __shared__ unsigned char smem_raw[];
  float2*         Zbuf  = (float2*)smem_raw;                    // [4][2048] complex f32  65536 B
  float2*         Ttab  = Zbuf + 4 * AC_L;                      // [64*32] twiddles f32   16384 B
  float*          topV  = (float*)(Ttab + 64 * 32);             // [2048*7]               57344 B
  unsigned short* F64rH = (unsigned short*)(topV + AC_L * AC_TOPK); // 4x [64*64] bf16    32768 B
  unsigned short* F64rL = F64rH + 64 * 64;
  unsigned short* F64iH = F64rL + 64 * 64;
  unsigned short* F64iL = F64iH + 64 * 64;
  unsigned short* F32rH = F64iL + 64 * 64;                      // 4x [32*32] bf16         8192 B
  unsigned short* F32rL = F32rH + 32 * 32;
  unsigned short* F32iH = F32rL + 32 * 32;
  unsigned short* F32iL = F32iH + 32 * 32;
  unsigned char*  topD  = (unsigned char*)(F32iL + 32 * 32);    // [2048*7] delays        14336 B

  const float PI2 = 6.283185307179586f;

  // ---------------- Build DFT / twiddle tables (forward only) ----------------
  for (int i = tid; i < 64 * 64; i += 256) {
    int r = i >> 6, c = i & 63;
    float s, co; __sincosf((float)((r * c) & 63) * (PI2 / 64.f), &s, &co);
    unsigned short hb;
    hb = f2bf(co); F64rH[i] = hb; F64rL[i] = f2bf(co - bf2f(hb));
    hb = f2bf(-s); F64iH[i] = hb; F64iL[i] = f2bf(-s - bf2f(hb));
  }
  for (int i = tid; i < 32 * 32; i += 256) {
    int r = i >> 5, c = i & 31;
    float s, co; __sincosf((float)((r * c) & 31) * (PI2 / 32.f), &s, &co);
    unsigned short hb;
    hb = f2bf(co); F32rH[i] = hb; F32rL[i] = f2bf(co - bf2f(hb));
    hb = f2bf(-s); F32iH[i] = hb; F32iL[i] = f2bf(-s - bf2f(hb));
  }
  for (int i = tid; i < 64 * 32; i += 256) {
    int n1 = i >> 5, k2 = i & 31;
    float s, co; __sincosf((float)((n1 * k2) & (AC_L - 1)) * (PI2 / (float)AC_L), &s, &co);
    Ttab[i] = make_float2(co, -s);
  }
  for (int i = tid; i < AC_L * AC_TOPK; i += 256) { topV[i] = -3.0e38f; topD[i] = 0; }
  __syncthreads();

  // ---------------- WMMA operand builders (contiguous k = e + 16*(lane>>4), same in A and B) ----------------
  // A 16x32 bf16: lane m=lane&15 is row M; K = e + 16*(lane>>4)
  auto loadA_data = [&](const float2* Zc, int m0, int plane, v16bf& H, v16bf& L) {
    V16U rh, rl; int m = lane & 15, kh = lane >> 4;
#pragma unroll
    for (int e = 0; e < 16; ++e) {
      int K = e + 16 * kh;
      float2 z = Zc[(m0 + m) + 64 * K];
      float f = plane ? z.y : z.x;
      unsigned short hb = f2bf(f);
      rh.u[e] = hb;
      rl.u[e] = f2bf(f - bf2f(hb));
    }
    H = rh.v; L = rl.v;
  };
  // B 32x16 bf16: lane n=lane&15 is col N; K = e + 16*(lane>>4)
  auto loadB_tab32 = [&](const unsigned short* tH, const unsigned short* tL,
                         int c0, bool neg, v16bf& H, v16bf& L) {
    V16U rh, rl; int n = lane & 15, kh = lane >> 4;
#pragma unroll
    for (int e = 0; e < 16; ++e) {
      int idx = (e + 16 * kh) * 32 + (c0 + n);
      unsigned short uh = tH[idx], ul = tL[idx];
      if (neg) { uh ^= 0x8000u; ul ^= 0x8000u; }   // -(hi+lo) = (-hi)+(-lo)
      rh.u[e] = uh; rl.u[e] = ul;
    }
    H = rh.v; L = rl.v;
  };
  auto loadA_tab64 = [&](const unsigned short* tH, const unsigned short* tL,
                         int m0, int k0, bool neg, v16bf& H, v16bf& L) {
    V16U rh, rl; int m = lane & 15, kh = lane >> 4;
#pragma unroll
    for (int e = 0; e < 16; ++e) {
      int K = k0 + e + 16 * kh;
      int idx = (m0 + m) * 64 + K;
      unsigned short uh = tH[idx], ul = tL[idx];
      if (neg) { uh ^= 0x8000u; ul ^= 0x8000u; }
      rh.u[e] = uh; rl.u[e] = ul;
    }
    H = rh.v; L = rl.v;
  };
  auto loadB_data = [&](const float2* Zc, int k0, int c0, int plane, v16bf& H, v16bf& L) {
    V16U rh, rl; int n = lane & 15, kh = lane >> 4;
#pragma unroll
    for (int e = 0; e < 16; ++e) {
      int K = k0 + e + 16 * kh;
      float2 z = Zc[K + 64 * (c0 + n)];
      float f = plane ? z.y : z.x;
      unsigned short hb = f2bf(f);
      rh.u[e] = hb;
      rl.u[e] = f2bf(f - bf2f(hb));
    }
    H = rh.v; L = rl.v;
  };

  // ---------------- Four-step forward FFT of length 2048, in LDS ----------------
  // n = n1 + 64*n2 ; X[32*k1+k2] = sum_n1 W64^{n1 k1} W2048^{n1 k2} (sum_n2 Z[n1+64 n2] W32^{n2 k2})
  auto fft2048 = [&](float2* Z0, int nch, int chstep) {
    // Step A: G = D * F32, in-place per 16-row band (band fully in regs first)
    for (int task = wave; task < nch * 4; task += 8) {
      int ch = task >> 2;
      int m0 = (task & 3) * 16;
      float2* Zc = Z0 + (size_t)(ch * chstep) * AC_L;
      v16bf ArH, ArL, AiH, AiL;
      loadA_data(Zc, m0, 0, ArH, ArL);
      loadA_data(Zc, m0, 1, AiH, AiL);
      v8f gr[2], gi[2];
#pragma unroll
      for (int nt = 0; nt < 2; ++nt) {
        int c0 = nt * 16;
        v16bf BrH, BrL, BiH, BiL, BnH, BnL;
        loadB_tab32(F32rH, F32rL, c0, false, BrH, BrL);
        loadB_tab32(F32iH, F32iL, c0, false, BiH, BiL);
        loadB_tab32(F32iH, F32iL, c0, true,  BnH, BnL);
        v8f ar = {0,0,0,0,0,0,0,0};
        ar = wmma3(ArH, ArL, BrH, BrL, ar);      //  Dr*F32r
        ar = wmma3(AiH, AiL, BnH, BnL, ar);      // -Di*F32i
        v8f ai = {0,0,0,0,0,0,0,0};
        ai = wmma3(ArH, ArL, BiH, BiL, ai);      //  Dr*F32i
        ai = wmma3(AiH, AiL, BrH, BrL, ai);      //  Di*F32r
        gr[nt] = ar; gi[nt] = ai;
      }
      int mh = lane >> 4, n = lane & 15;
#pragma unroll
      for (int nt = 0; nt < 2; ++nt) {
        int c0 = nt * 16;
#pragma unroll
        for (int e = 0; e < 8; ++e) {
          int n1 = m0 + e + 8 * mh;
          Zc[n1 + 64 * (c0 + n)] = make_float2(gr[nt][e], gi[nt][e]);
        }
      }
    }
    __syncthreads();
    // Step B: twiddle G' = G .* W2048^{n1 k2}  (exact f32, pointwise)
    for (int i = tid; i < nch * AC_L; i += 256) {
      int ch = i >> 11, p = i & (AC_L - 1);
      float2* Zc = Z0 + (size_t)(ch * chstep) * AC_L;
      int n1 = p & 63, k2 = p >> 6;
      float2 t = Ttab[n1 * 32 + k2];
      float2 z = Zc[p];
      Zc[p] = make_float2(z.x * t.x - z.y * t.y, z.x * t.y + z.y * t.x);
    }
    __syncthreads();
    // Step C: X = F64 * G'; output scattered to natural order 32*k1+k2.
    // Pull all B operands into regs, barrier, then compute+scatter (no races).
    bool active = wave < nch * 2;
    int ch = wave >> 1, c0 = (wave & 1) * 16;
    float2* Zc = Z0 + (size_t)(ch * chstep) * AC_L;
    v16bf Br0H, Br0L, Bi0H, Bi0L, Br1H, Br1L, Bi1H, Bi1L;
    if (active) {
      loadB_data(Zc,  0, c0, 0, Br0H, Br0L);
      loadB_data(Zc,  0, c0, 1, Bi0H, Bi0L);
      loadB_data(Zc, 32, c0, 0, Br1H, Br1L);
      loadB_data(Zc, 32, c0, 1, Bi1H, Bi1L);
    }
    __syncthreads();
    if (active) {
      int mh = lane >> 4, n = lane & 15;
      for (int mt = 0; mt < 4; ++mt) {
        int m0 = mt * 16;
        // phase 1: A = F64r
        v16bf A0H, A0L, A1H, A1L;
        loadA_tab64(F64rH, F64rL, m0, 0,  false, A0H, A0L);
        loadA_tab64(F64rH, F64rL, m0, 32, false, A1H, A1L);
        v8f xr = {0,0,0,0,0,0,0,0};
        xr = wmma3(A0H, A0L, Br0H, Br0L, xr);    //  Fr*Gr
        xr = wmma3(A1H, A1L, Br1H, Br1L, xr);
        v8f xi = {0,0,0,0,0,0,0,0};
        xi = wmma3(A0H, A0L, Bi0H, Bi0L, xi);    //  Fr*Gi
        xi = wmma3(A1H, A1L, Bi1H, Bi1L, xi);
        // phase 2: A = -F64i
        loadA_tab64(F64iH, F64iL, m0, 0,  true, A0H, A0L);
        loadA_tab64(F64iH, F64iL, m0, 32, true, A1H, A1L);
        xr = wmma3(A0H, A0L, Bi0H, Bi0L, xr);    // -Fi*Gi
        xr = wmma3(A1H, A1L, Bi1H, Bi1L, xr);
        v8f t = {0,0,0,0,0,0,0,0};
        t = wmma3(A0H, A0L, Br0H, Br0L, t);      // -Fi*Gr
        t = wmma3(A1H, A1L, Br1H, Br1L, t);
#pragma unroll
        for (int e = 0; e < 8; ++e) {
          int k1 = m0 + e + 8 * mh;
          Zc[32 * k1 + (c0 + n)] = make_float2(xr[e], xi[e] - t[e]); // xi_final = Fr*Gi + Fi*Gr
        }
      }
    }
    __syncthreads();
  };

  // ---------------- Main loop: 16 blocks of 4 channels ----------------
  const size_t qkBase = ((size_t)b * AC_L) * AC_E + (size_t)h * AC_HD;
  for (int cb = 0; cb < 16; ++cb) {
    const int cbase = cb * 4;
    // Load Z[c] = q + i*k for 4 channels (16B loads; L2 amortizes line reuse
    // across the 8 heads of the same batch running concurrently)
    for (int n = tid; n < AC_L; n += 256) {
      const float4 qv = *(const float4*)(Qg + qkBase + (size_t)n * AC_E + cbase);
      const float4 kv = *(const float4*)(Kg + qkBase + (size_t)n * AC_E + cbase);
      Zbuf[0 * AC_L + n] = make_float2(qv.x, kv.x);
      Zbuf[1 * AC_L + n] = make_float2(qv.y, kv.y);
      Zbuf[2 * AC_L + n] = make_float2(qv.z, kv.z);
      Zbuf[3 * AC_L + n] = make_float2(qv.w, kv.w);
    }
    __syncthreads();

    fft2048(Zbuf, 4, 1);                       // Zhat = FFT(q + i k), 4 channels

    // P[f] = Qhat[f] * conj(Khat[f]), unpacked via Hermitian symmetry; in place.
    for (int i = tid; i < 4 * 1025; i += 256) {
      int ch = i / 1025, f = i - ch * 1025;
      float2* Zc = Zbuf + (size_t)ch * AC_L;
      int f2 = (AC_L - f) & (AC_L - 1);
      float2 z1 = Zc[f], z2 = Zc[f2];
      float qr = 0.5f * (z1.x + z2.x), qi = 0.5f * (z1.y - z2.y);
      float kr = 0.5f * (z1.y + z2.y), ki = 0.5f * (z2.x - z1.x);
      float pr = qr * kr + qi * ki;
      float pi = qi * kr - qr * ki;
      Zc[f]  = make_float2(pr,  pi);
      Zc[f2] = make_float2(pr, -pi);           // Hermitian; benign overwrite at f==f2
    }
    __syncthreads();

    // Pack Wc = conj(P_even + i*P_odd) into even buffers; IFFT = conj(FFT(Wc))/N
    for (int i = tid; i < 2 * AC_L; i += 256) {
      int pr_ = i >> 11, f = i & (AC_L - 1);
      float2* Za = Zbuf + (size_t)(pr_ * 2) * AC_L;
      float2* Zb = Zbuf + (size_t)(pr_ * 2 + 1) * AC_L;
      float2 a = Za[f], bb = Zb[f];
      Za[f] = make_float2(a.x - bb.y, -(a.y + bb.x));
    }
    __syncthreads();

    fft2048(Zbuf, 2, 2);                       // buffers 0 and 2

    // corr for channels cbase..cbase+3; update running top-7 per position l
    const float sc = 1.0f / (float)AC_L;
    for (int l = tid; l < AC_L; l += 256) {
      float2 u = Zbuf[l];
      float2 w = Zbuf[2 * AC_L + l];
      float cv[4] = { u.x * sc, -u.y * sc, w.x * sc, -w.y * sc };
      float* tv = topV + l * AC_TOPK;
      unsigned char* td = topD + l * AC_TOPK;
#pragma unroll
      for (int c = 0; c < 4; ++c) {
        float val = cv[c];
        int pos = AC_TOPK;
        while (pos > 0 && val > tv[pos - 1]) pos--;
        if (pos < AC_TOPK) {
          for (int j = AC_TOPK - 1; j > pos; --j) { tv[j] = tv[j - 1]; td[j] = td[j - 1]; }
          tv[pos] = val; td[pos] = (unsigned char)(cbase + c);
        }
      }
    }
    __syncthreads();
  }

  // ---------------- Softmax + delayed-V aggregation ----------------
  const size_t vBase = ((size_t)b * AC_L) * AC_E + (size_t)h * AC_HD;
  for (int l = wave; l < AC_L; l += 8) {
    const float* tv = topV + l * AC_TOPK;
    const unsigned char* td = topD + l * AC_TOPK;
    float wj[AC_TOPK];
    float mx = tv[0], s = 0.f;
#pragma unroll
    for (int j = 0; j < AC_TOPK; ++j) { wj[j] = __expf(tv[j] - mx); s += wj[j]; }
    float inv = 1.0f / s;
    float a0 = 0.f, a1 = 0.f;
    // prefetch the first delayed row, then software-pipeline the rest
    {
      int row0 = (l - (int)td[0]) & (AC_L - 1);
      __builtin_prefetch(Vg + vBase + (size_t)row0 * AC_E, 0, 3);
    }
#pragma unroll
    for (int j = 0; j < AC_TOPK; ++j) {
      int row = (l - (int)td[j]) & (AC_L - 1);
      const float* vp = Vg + vBase + (size_t)row * AC_E;
      if (j + 1 < AC_TOPK) {
        int rown = (l - (int)td[j + 1]) & (AC_L - 1);
        __builtin_prefetch(Vg + vBase + (size_t)rown * AC_E, 0, 3);
      }
      a0 += wj[j] * vp[lane];
      a1 += wj[j] * vp[lane + 32];
    }
    float* op = Og + ((size_t)b * AC_L + l) * AC_E + (size_t)h * AC_HD;
    op[lane]      = a0 * inv;
    op[lane + 32] = a1 * inv;
  }
}

// ---------------------------------------------------------------------------
extern "C" void kernel_launch(void* const* d_in, const int* in_sizes, int n_in,
                              void* d_out, int out_size, void* d_ws, size_t ws_size,
                              hipStream_t stream) {
  (void)in_sizes; (void)n_in; (void)out_size; (void)d_ws; (void)ws_size;
  const float* Qg = (const float*)d_in[0];
  const float* Kg = (const float*)d_in[1];
  const float* Vg = (const float*)d_in[2];
  float* Og = (float*)d_out;

  constexpr size_t LDS_BYTES =
      (size_t)4 * AC_L * sizeof(float2)              // Zbuf
    + (size_t)64 * 32 * sizeof(float2)               // Ttab
    + (size_t)AC_L * AC_TOPK * sizeof(float)         // topV
    + (size_t)64 * 64 * 4 * sizeof(unsigned short)   // F64 re/im hi/lo
    + (size_t)32 * 32 * 4 * sizeof(unsigned short)   // F32 re/im hi/lo
    + (size_t)AC_L * AC_TOPK;                        // topD
  // = 194560 bytes, fits CDNA5's 320KB-per-workgroup LDS (1 WG per WGP).

  autocorr_fft_topk_kernel<<<dim3(AC_B * AC_H), dim3(256), LDS_BYTES, stream>>>(Qg, Kg, Vg, Og);
}
